// ASTPathsEncoder_343597383863
// MI455X (gfx1250) — matrix operand order repaired and checked by reading, not compiled, not run on hardware
//
#include <hip/hip_runtime.h>
#include <hip/hip_bf16.h>

#define P_ 2048
#define L_ 32
#define D_ 512
#define N_NODES_ 20000

typedef __attribute__((ext_vector_type(16))) __bf16 v16bf;
typedef __attribute__((ext_vector_type(8)))  __bf16 v8bf;
typedef __attribute__((ext_vector_type(8)))  float  v8f;

// ---------- fragment loaders (16-bit WMMA VGPR layouts, CDNA5 ISA) ----------
// A 16x32 (MxK): lane<16 -> M=lane, halfs0-7=K0..7, halfs8-15=K16..23
//                lane>=16 -> M=lane-16, halfs0-7=K8..15, halfs8-15=K24..31
// Caller passes p = rowbase + (lane<16?0:8); we load [p..p+7] and [p+16..p+23].
__device__ __forceinline__ v16bf loadA16(const __bf16* p) {
  union { v16bf v; v8bf h[2]; } u;
  u.h[0] = *(const v8bf*)(p);
  u.h[1] = *(const v8bf*)(p + 16);
  return u.v;
}
// B 32x16 (KxN), weights stored transposed [n][k]: lane<16 -> N=lane, K=0..7,16..23;
// lane>=16 -> N=lane-16, K=8..15,24..31.  Caller passes p = WT + n*512 + (lane<16?0:8).
__device__ __forceinline__ v16bf loadB16(const __bf16* p) {
  union { v16bf v; v8bf h[2]; } u;
  u.h[0] = *(const v8bf*)(p);
  u.h[1] = *(const v8bf*)(p + 16);
  return u.v;
}
#define WMMA_BF16(a,b,c) __builtin_amdgcn_wmma_f32_16x16x32_bf16(false,(a),false,(b),(short)0,(c),false,false)

// ---------- small prep kernels ----------
__global__ void k_zero(float* __restrict__ p, int n) {
  for (int i = blockIdx.x * blockDim.x + threadIdx.x; i < n; i += gridDim.x * blockDim.x)
    p[i] = 0.f;
}

// OA[v][d] = sum_k orient_emb[v][k]*proj_W[k][d] + proj_b[d]
// OB[v][d] = sum_k orient_emb[v][k]*proj_W[512+k][d]
__global__ void k_orient_proj(const float* __restrict__ oe, const float* __restrict__ W,
                              const float* __restrict__ b, float* __restrict__ OA,
                              float* __restrict__ OB) {
  int i = blockIdx.x * 256 + threadIdx.x;      // 0..8191
  if (i >= 2 * 8 * D_) return;
  int which = i >> 12;                          // 0=OA, 1=OB
  int v = (i >> 9) & 7;
  int d = i & (D_ - 1);
  float s = which ? 0.f : b[d];
  const float* wbase = W + (size_t)(which * D_) * D_ + d;
  const float* ebase = oe + (size_t)v * D_;
  for (int k = 0; k < D_; ++k) s += ebase[k] * wbase[(size_t)k * D_];
  (which ? OB : OA)[v * D_ + d] = s;
}

// Transpose + convert weights to bf16 [3D][D] (row = gate column n, contiguous k)
__global__ void k_prep_weights(const float* __restrict__ Wx, const float* __restrict__ Wh,
                               __bf16* __restrict__ WxT, __bf16* __restrict__ WhT) {
  int i = blockIdx.x * 256 + threadIdx.x;       // over 1536*512
  if (i >= 3 * D_ * D_) return;
  int n = i / D_, k = i - n * D_;
  WxT[i] = (__bf16)Wx[(size_t)k * (3 * D_) + n];
  WhT[i] = (__bf16)Wh[(size_t)k * (3 * D_) + n];
}

// Build woven input X (bf16, [P][64][D]): t=2l node embedding, t=2l+1 orientation, masked.
__global__ void k_build_x(const int* __restrict__ node_idx, const int* __restrict__ lengths,
                          const int* __restrict__ child, const int* __restrict__ vert,
                          const int* __restrict__ ntypes, const float* __restrict__ type_emb,
                          const float* __restrict__ OA, const float* __restrict__ OB,
                          __bf16* __restrict__ Xb) {
  long long idx = (long long)blockIdx.x * 256 + threadIdx.x;   // P*L*D
  if (idx >= (long long)P_ * L_ * D_) return;
  int d = (int)(idx & (D_ - 1));
  int pl = (int)(idx >> 9);
  int l = pl & (L_ - 1);
  int p = pl >> 5;
  __bf16 xe = (__bf16)0.f, xo = (__bf16)0.f;
  if (l < lengths[p]) {
    int ni = node_idx[pl];
    int ty = ntypes[ni];
    xe = (__bf16)type_emb[(size_t)ty * D_ + d];
    int cp = child[pl], vd = vert[pl];
    xo = (__bf16)(OA[cp * D_ + d] + OB[vd * D_ + d]);
  }
  size_t base = ((size_t)p * 64 + 2 * l) * D_ + d;
  Xb[base] = xe;
  Xb[base + D_] = xo;
}

// ---------- GRU step epilogue helper ----------
__device__ __forceinline__ void epilogue16(v8f ar, v8f az, v8f anx, v8f anh,
    int rbase, int col, float bxr, float bhr, float bxz, float bhz, float bxn, float bhn,
    const float* __restrict__ Hf_in, float* __restrict__ Hf_out, __bf16* __restrict__ Hb_out,
    const int* __restrict__ lengths, float* __restrict__ outArr, int l) {
#pragma unroll
  for (int i = 0; i < 8; ++i) {
    int row = rbase + i;
    float r = 1.f / (1.f + __expf(-(ar[i] + bxr + bhr)));
    float z = 1.f / (1.f + __expf(-(az[i] + bxz + bhz)));
    float n = tanhf(anx[i] + bxn + r * (anh[i] + bhn));
    size_t hidx = (size_t)row * D_ + col;
    float hp = Hf_in[hidx];
    bool m = l < lengths[row];
    float hn = m ? ((1.f - z) * n + z * hp) : hp;
    Hf_out[hidx] = hn;
    Hb_out[hidx] = (__bf16)hn;
    outArr[(size_t)row * (L_ * D_) + (size_t)l * D_ + col] = m ? hn : 0.f;
  }
}

// ---------- GRU step: gates = [x_t;h] @ [Wx;Wh], bf16 WMMA, f32 accum ----------
// Block: 256 threads = 8 waves, arranged 4(M) x 2(N); each wave does two 16-row
// subtiles -> block tile 128 rows x 32 cols of H. Grid (512/32, 2048/128).
__global__ __launch_bounds__(256)
void k_gru_step(const __bf16* __restrict__ Xb, const __bf16* __restrict__ Hb_in,
                const float* __restrict__ Hf_in, float* __restrict__ Hf_out,
                __bf16* __restrict__ Hb_out, const __bf16* __restrict__ WxT,
                const __bf16* __restrict__ WhT, const float* __restrict__ bx,
                const float* __restrict__ bh, const int* __restrict__ lengths,
                float* __restrict__ out_nodes, float* __restrict__ out_orient, int t) {
  const int lane  = threadIdx.x & 31;
  const int wave  = threadIdx.x >> 5;
  const int wm    = wave >> 1;               // 0..3
  const int wn    = wave & 1;                // 0..1
  const int mBase = blockIdx.y * 128 + wm * 32;
  const int n0    = blockIdx.x * 32 + wn * 16;
  const int lhalf = (lane < 16) ? 0 : 1;
  const int lcol  = lane & 15;

  const int kA = lhalf ? 8 : 0;              // A-fragment K base offset
  const int kB = lhalf ? 8 : 0;              // B-fragment K base offset
  const __bf16* Xt = Xb + (size_t)t * D_;    // row stride 64*D for fixed t

  const __bf16* xp0 = Xt + (size_t)(mBase + lcol) * (64 * D_) + kA;
  const __bf16* xp1 = Xt + (size_t)(mBase + 16 + lcol) * (64 * D_) + kA;
  const __bf16* hp0 = Hb_in + (size_t)(mBase + lcol) * D_ + kA;
  const __bf16* hp1 = Hb_in + (size_t)(mBase + 16 + lcol) * D_ + kA;
  const int nrow = n0 + lcol;
  const __bf16* pbxr = WxT + (size_t)nrow * D_ + kB;
  const __bf16* pbxz = WxT + (size_t)(D_ + nrow) * D_ + kB;
  const __bf16* pbxn = WxT + (size_t)(2 * D_ + nrow) * D_ + kB;
  const __bf16* pbhr = WhT + (size_t)nrow * D_ + kB;
  const __bf16* pbhz = WhT + (size_t)(D_ + nrow) * D_ + kB;
  const __bf16* pbhn = WhT + (size_t)(2 * D_ + nrow) * D_ + kB;

  v8f accr0 = {}, accz0 = {}, accnx0 = {}, accnh0 = {};
  v8f accr1 = {}, accz1 = {}, accnx1 = {}, accnh1 = {};

  for (int k0 = 0; k0 < D_; k0 += 32) {
    v16bf ax0 = loadA16(xp0 + k0);
    v16bf ax1 = loadA16(xp1 + k0);
    v16bf ah0 = loadA16(hp0 + k0);
    v16bf ah1 = loadA16(hp1 + k0);
    v16bf b;
    b = loadB16(pbxr + k0); accr0  = WMMA_BF16(ax0, b, accr0);  accr1  = WMMA_BF16(ax1, b, accr1);
    b = loadB16(pbhr + k0); accr0  = WMMA_BF16(ah0, b, accr0);  accr1  = WMMA_BF16(ah1, b, accr1);
    b = loadB16(pbxz + k0); accz0  = WMMA_BF16(ax0, b, accz0);  accz1  = WMMA_BF16(ax1, b, accz1);
    b = loadB16(pbhz + k0); accz0  = WMMA_BF16(ah0, b, accz0);  accz1  = WMMA_BF16(ah1, b, accz1);
    b = loadB16(pbxn + k0); accnx0 = WMMA_BF16(ax0, b, accnx0); accnx1 = WMMA_BF16(ax1, b, accnx1);
    b = loadB16(pbhn + k0); accnh0 = WMMA_BF16(ah0, b, accnh0); accnh1 = WMMA_BF16(ah1, b, accnh1);
  }

  const int col = n0 + lcol;
  const float bxr = bx[col],          bhr = bh[col];
  const float bxz = bx[D_ + col],     bhz = bh[D_ + col];
  const float bxn = bx[2 * D_ + col], bhn = bh[2 * D_ + col];
  const int l = t >> 1;
  float* outArr = (t & 1) ? out_orient : out_nodes;
  // D-matrix layout: VGPR i -> M = i + (lane<16?0:8)
  epilogue16(accr0, accz0, accnx0, accnh0, mBase + lhalf * 8, col,
             bxr, bhr, bxz, bhz, bxn, bhn, Hf_in, Hf_out, Hb_out, lengths, outArr, l);
  epilogue16(accr1, accz1, accnx1, accnh1, mBase + 16 + lhalf * 8, col,
             bxr, bhr, bxz, bhz, bxn, bhn, Hf_in, Hf_out, Hb_out, lengths, outArr, l);
}

// ---------- segment sum: one block per (p,l) row, fp32 global atomics ----------
__global__ void k_segment(const float* __restrict__ nodes_enc, const int* __restrict__ node_idx,
                          float* __restrict__ seg) {
  int pl = blockIdx.x;                           // 0..P*L-1
  int target = node_idx[pl];
  const float* src = nodes_enc + (size_t)pl * D_;
  float* dst = seg + (size_t)target * D_;
  for (int j = threadIdx.x; j < D_; j += blockDim.x) {
    float v = src[j];
    if (v != 0.f)
      __hip_atomic_fetch_add(&dst[j], v, __ATOMIC_RELAXED, __HIP_MEMORY_SCOPE_AGENT);
  }
}

extern "C" void kernel_launch(void* const* d_in, const int* in_sizes, int n_in,
                              void* d_out, int out_size, void* d_ws, size_t ws_size,
                              hipStream_t stream) {
  const int*   node_idx  = (const int*)d_in[0];
  const int*   lengths   = (const int*)d_in[1];
  /* d_in[2] = mask: recomputed from lengths instead */
  const int*   child     = (const int*)d_in[3];
  const int*   vert      = (const int*)d_in[4];
  const int*   ntypes    = (const int*)d_in[5];
  const float* type_emb  = (const float*)d_in[6];
  const float* orientemb = (const float*)d_in[7];
  const float* projW     = (const float*)d_in[8];
  const float* projb     = (const float*)d_in[9];
  const float* Wx        = (const float*)d_in[10];
  const float* Wh        = (const float*)d_in[11];
  const float* bx        = (const float*)d_in[12];
  const float* bh        = (const float*)d_in[13];

  float* seg        = (float*)d_out;                       // [20000][512]
  float* nodes_enc  = seg + (size_t)N_NODES_ * D_;         // [2048][32][512]
  float* orient_enc = nodes_enc + (size_t)P_ * L_ * D_;    // [2048][32][512]

  char* ws = (char*)d_ws;
  size_t off = 0;
  auto alloc = [&](size_t bytes) { void* p = ws + off; off += (bytes + 255) & ~(size_t)255; return p; };
  float*  OA   = (float*)alloc(8 * D_ * sizeof(float));
  float*  OB   = (float*)alloc(8 * D_ * sizeof(float));
  __bf16* WxT  = (__bf16*)alloc((size_t)3 * D_ * D_ * 2);
  __bf16* WhT  = (__bf16*)alloc((size_t)3 * D_ * D_ * 2);
  float*  Hf0  = (float*)alloc((size_t)P_ * D_ * 4);
  float*  Hf1  = (float*)alloc((size_t)P_ * D_ * 4);
  __bf16* Hb0  = (__bf16*)alloc((size_t)P_ * D_ * 2);
  __bf16* Hb1  = (__bf16*)alloc((size_t)P_ * D_ * 2);
  __bf16* Xb   = (__bf16*)alloc((size_t)P_ * 64 * D_ * 2);   // ~128 MB
  (void)ws_size; (void)in_sizes; (void)n_in; (void)out_size;

  // prep
  k_orient_proj<<<32, 256, 0, stream>>>(orientemb, projW, projb, OA, OB);
  k_prep_weights<<<(3 * D_ * D_ + 255) / 256, 256, 0, stream>>>(Wx, Wh, WxT, WhT);
  k_zero<<<4096, 256, 0, stream>>>(seg, N_NODES_ * D_);
  k_zero<<<2048, 256, 0, stream>>>(Hf0, P_ * D_);
  k_zero<<<1024, 256, 0, stream>>>((float*)Hb0, P_ * D_ / 2);
  k_build_x<<<(int)(((long long)P_ * L_ * D_ + 255) / 256), 256, 0, stream>>>(
      node_idx, lengths, child, vert, ntypes, type_emb, OA, OB, Xb);

  // 64 sequential GRU steps (ping-pong H)
  dim3 grid(D_ / 32, P_ / 128);
  for (int t = 0; t < 64; ++t) {
    const float*  hfi = (t & 1) ? Hf1 : Hf0;
    float*        hfo = (t & 1) ? Hf0 : Hf1;
    const __bf16* hbi = (t & 1) ? Hb1 : Hb0;
    __bf16*       hbo = (t & 1) ? Hb0 : Hb1;
    k_gru_step<<<grid, 256, 0, stream>>>(Xb, hbi, hfi, hfo, hbo, WxT, WhT,
                                         bx, bh, lengths, nodes_enc, orient_enc, t);
  }

  // segment sum of masked node encodings
  k_segment<<<P_ * L_, 256, 0, stream>>>(nodes_enc, node_idx, seg);
}
